// MixtureOfExperts_46462956208737
// MI455X (gfx1250) — matrix operand order repaired and checked by reading, not compiled, not run on hardware
//
#include <hip/hip_runtime.h>
#include <math.h>

// ---------------------------------------------------------------------------
// MoE (top-2 of 8 experts), D=1024, D_FF=4096, N=8192 tokens.
// bf16 WMMA GEMMs (v_wmma_f32_16x16x32_bf16), routed (top-2 only) compute.
// Double-buffered LDS, register-prefetched staging, fragment-preswizzled weights.
// ---------------------------------------------------------------------------

typedef __attribute__((ext_vector_type(16))) __bf16 v16bf;
typedef __attribute__((ext_vector_type(8)))  __bf16 v8bf;
typedef __attribute__((ext_vector_type(8)))  float  v8f;
typedef unsigned short u16;
typedef unsigned int   u32;
typedef __attribute__((ext_vector_type(8))) u16 v8u16;

static constexpr int kD = 1024;   // d_model
static constexpr int kF = 4096;   // d_ff
static constexpr int kE = 8;      // experts
static constexpr int kN = 8192;   // tokens (4*2048)

union FragV { v16bf v; v8bf h[2]; };

__device__ __forceinline__ u16 f2bf(float f) {
  union { float f; u32 u; } c; c.f = f;
  u32 u = c.u;
  return (u16)((u + 0x7FFFu + ((u >> 16) & 1u)) >> 16);   // RNE
}

__device__ __forceinline__ v8f wmma_bf16(v16bf a, v16bf b, v8f c) {
  return __builtin_amdgcn_wmma_f32_16x16x32_bf16(false, a, false, b, (short)0, c, false, false);
}

// ---------------------------------------------------------------------------
// fp32 -> bf16 linear conversion (for x), 8 elements / thread
// ---------------------------------------------------------------------------
__global__ __launch_bounds__(256) void cvt_bf16_kernel(const float* __restrict__ src,
                                                       u16* __restrict__ dst, int n) {
  size_t i = ((size_t)blockIdx.x * 256 + threadIdx.x) * 8;
  if (i + 8 > (size_t)n) return;
  float4 a = *(const float4*)(src + i);
  float4 b = *(const float4*)(src + i + 4);
  v8u16 o;
  o[0] = f2bf(a.x); o[1] = f2bf(a.y); o[2] = f2bf(a.z); o[3] = f2bf(a.w);
  o[4] = f2bf(b.x); o[5] = f2bf(b.y); o[6] = f2bf(b.z); o[7] = f2bf(b.w);
  *(v8u16*)(dst + i) = o;
}

// ---------------------------------------------------------------------------
// fp32 -> bf16 weight conversion into fragment-swizzled layout:
//   Wp[((e*KB + kb)*(ldn/16) + nsub)*512 + lane*16 + elem]
// where kb = k/32, nsub = n/16, lane = (n%16) + 16*((k%32)>=16), elem = k%16.
// GEMM B-staging then becomes a contiguous 8KB copy per 32x128 tile.
// ---------------------------------------------------------------------------
__global__ __launch_bounds__(256) void swz_w_kernel(const float* __restrict__ W,
                                                    u16* __restrict__ Wp,
                                                    int ldn, int KB) {
  __shared__ __align__(16) u16 S[4096];
  const int tid = threadIdx.x;
  const int ntile = blockIdx.x, kb = blockIdx.y, e = blockIdx.z;
  const float* src = W + ((size_t)e * KB + kb) * 32 * ldn + ntile * 128;
  for (int c = tid; c < 512; c += 256) {
    const int k = c >> 4;
    const int n = (c & 15) * 8;
    float4 f0 = *(const float4*)(src + (size_t)k * ldn + n);
    float4 f1 = *(const float4*)(src + (size_t)k * ldn + n + 4);
    u16 q[8] = { f2bf(f0.x), f2bf(f0.y), f2bf(f0.z), f2bf(f0.w),
                 f2bf(f1.x), f2bf(f1.y), f2bf(f1.z), f2bf(f1.w) };
    const int sub = n >> 4;
    const int lb  = (n & 15) + ((k & 16) ? 16 : 0);
    u16* dst = &S[((sub * 32 + lb) << 4) + (k & 15)];
#pragma unroll
    for (int j = 0; j < 8; ++j) dst[j * 16] = q[j];
  }
  __syncthreads();
  u16* dstg = Wp + (((size_t)e * KB + kb) * (ldn / 16) + ntile * 8) * 512 + tid * 16;
  *(v8u16*)dstg       = *(v8u16*)&S[tid * 16];
  *(v8u16*)(dstg + 8) = *(v8u16*)&S[tid * 16 + 8];
}

// ---------------------------------------------------------------------------
// Gating: logits = x @ Wg, softmax, top-2, compact per-expert token lists.
// ---------------------------------------------------------------------------
__global__ __launch_bounds__(256) void gate_kernel(const float* __restrict__ x,
                                                   const float* __restrict__ Wg,
                                                   int*   __restrict__ cnt,
                                                   int*   __restrict__ atok,
                                                   float* __restrict__ awgt,
                                                   float* __restrict__ probs_all) {
  __shared__ float wgs[kD * kE];          // 32 KB
  const int tid = threadIdx.x;
  for (int i = tid; i < kD * kE; i += 256) wgs[i] = Wg[i];
  __syncthreads();

  const int lane  = tid & 31;
  const int wave  = tid >> 5;
  const int token = blockIdx.x * 8 + wave;

  float acc[kE];
#pragma unroll
  for (int e = 0; e < kE; ++e) acc[e] = 0.0f;

  const float* xr = x + (size_t)token * kD;
  for (int k = lane; k < kD; k += 32) {
    const float xv = xr[k];
#pragma unroll
    for (int e = 0; e < kE; ++e) acc[e] += xv * wgs[k * kE + e];
  }
#pragma unroll
  for (int e = 0; e < kE; ++e)
    for (int off = 16; off; off >>= 1) acc[e] += __shfl_xor(acc[e], off, 32);

  if (lane == 0) {
    float m = acc[0];
#pragma unroll
    for (int e = 1; e < kE; ++e) m = fmaxf(m, acc[e]);
    float p[kE]; float s = 0.0f;
#pragma unroll
    for (int e = 0; e < kE; ++e) { p[e] = expf(acc[e] - m); s += p[e]; }
    const float inv = 1.0f / s;
#pragma unroll
    for (int e = 0; e < kE; ++e) { p[e] *= inv; probs_all[token * kE + e] = p[e]; }

    int i0 = 0; float v0 = p[0];
#pragma unroll
    for (int e = 1; e < kE; ++e) if (p[e] > v0) { v0 = p[e]; i0 = e; }
    int i1 = -1; float v1 = -1.0f;
#pragma unroll
    for (int e = 0; e < kE; ++e) if (e != i0 && p[e] > v1) { v1 = p[e]; i1 = e; }

    const float wsum = v0 + v1 + 1e-9f;
    const float w0 = v0 / wsum, w1 = v1 / wsum;

    int p0 = atomicAdd(&cnt[i0], 1);
    atok[i0 * kN + p0] = token; awgt[i0 * kN + p0] = w0;
    int p1 = atomicAdd(&cnt[i1], 1);
    atok[i1 * kN + p1] = token; awgt[i1 * kN + p1] = w1;
  }
}

// ---------------------------------------------------------------------------
// WMMA fragment loads from LDS (contiguous k = 16*(lane/16) + elem order in both A and B)
// ---------------------------------------------------------------------------
__device__ __forceinline__ v16bf load_a_frag(const u16* As /*[128][32]*/, int rowBase, int lane) {
  const int row = rowBase + (lane & 15);
  const int kb  = (lane < 16) ? 0 : 16;
  const u16* p = As + row * 32 + kb;
  FragV fa;
  fa.h[0] = *(const v8bf*)(p);
  fa.h[1] = *(const v8bf*)(p + 8);
  return fa.v;
}

__device__ __forceinline__ v16bf load_b_frag(const u16* Bs /*[8][32][16] swizzled*/, int sub, int lane) {
  const u16* p = Bs + ((sub * 32 + lane) << 4);
  FragV fb;
  fb.h[0] = *(const v8bf*)(p);
  fb.h[1] = *(const v8bf*)(p + 8);
  return fb.v;
}

// ---------------------------------------------------------------------------
// FFN1 per expert: h = GELU(x[tok] @ W1[e] + b1[e])  (bf16 out)
// Block 128x128, 8 waves as 4(M)x2(N), wave tile 32x64, BK=32,
// double-buffered LDS + register prefetch; B tiles pre-swizzled in global.
// ---------------------------------------------------------------------------
__global__ __launch_bounds__(256) void ffn1_kernel(const u16* __restrict__ xbf,
                                                   const u16* __restrict__ w1s,
                                                   const float* __restrict__ b1,
                                                   const int* __restrict__ cnt,
                                                   const int* __restrict__ atok,
                                                   u16* __restrict__ h, int expert) {
  const int count = cnt[expert];
  const int m0 = blockIdx.y * 128;
  if (m0 >= count) return;
  const int n0 = blockIdx.x * 128;

  __shared__ __align__(16) u16 As[2][128 * 32];
  __shared__ __align__(16) u16 Bs[2][4096];

  const int tid = threadIdx.x, lane = tid & 31, wave = tid >> 5;
  const int wm = wave >> 1, wn = wave & 1;

  // A staging: rows ar and ar+64, chunk column ak (K-invariant token gather)
  const int ar = tid >> 2;
  const int ak = (tid & 3) * 8;
  const int* tok = atok + expert * kN;
  const int mr0 = m0 + ar, mr1 = m0 + ar + 64;
  const int t0 = (mr0 < count) ? tok[mr0] : tok[0];
  const int t1 = (mr1 < count) ? tok[mr1] : tok[0];
  const u16* pA0 = xbf + (size_t)t0 * kD + ak;
  const u16* pA1 = xbf + (size_t)t1 * kD + ak;
  const int dA0 = ar * 32 + ak, dA1 = (ar + 64) * 32 + ak;

  // B staging: contiguous 8KB tile per K-step from pre-swizzled weights
  const u16* pB = w1s + (((size_t)expert * (kD / 32)) * (kF / 16) + (n0 >> 4)) * 512 + tid * 16;
  const size_t bStep = (size_t)(kF / 16) * 512;
  const int dB = tid * 16;

  v8f acc[2][4] = {};

  v8bf ra0 = *(const v8bf*)pA0;
  v8bf ra1 = *(const v8bf*)pA1;
  v8bf rb0 = *(const v8bf*)pB;
  v8bf rb1 = *(const v8bf*)(pB + 8);
  *(v8bf*)&As[0][dA0] = ra0;
  *(v8bf*)&As[0][dA1] = ra1;
  *(v8bf*)&Bs[0][dB]     = rb0;
  *(v8bf*)&Bs[0][dB + 8] = rb1;
  __syncthreads();

  const int KSTEPS = kD / 32;
  for (int it = 0; it < KSTEPS; ++it) {
    const int cur = it & 1, nxt = cur ^ 1;
    if (it + 1 < KSTEPS) {
      pA0 += 32; pA1 += 32; pB += bStep;
      ra0 = *(const v8bf*)pA0;
      ra1 = *(const v8bf*)pA1;
      rb0 = *(const v8bf*)pB;
      rb1 = *(const v8bf*)(pB + 8);
    }
    const u16* Ac = As[cur];
    const u16* Bc = Bs[cur];
    const v16bf a0 = load_a_frag(Ac, wm * 32, lane);
    const v16bf a1 = load_a_frag(Ac, wm * 32 + 16, lane);
    v16bf bf[4];
#pragma unroll
    for (int j = 0; j < 4; ++j) bf[j] = load_b_frag(Bc, wn * 4 + j, lane);
#pragma unroll
    for (int j = 0; j < 4; ++j) {
      acc[0][j] = wmma_bf16(a0, bf[j], acc[0][j]);
      acc[1][j] = wmma_bf16(a1, bf[j], acc[1][j]);
    }
    if (it + 1 < KSTEPS) {
      *(v8bf*)&As[nxt][dA0] = ra0;
      *(v8bf*)&As[nxt][dA1] = ra1;
      *(v8bf*)&Bs[nxt][dB]     = rb0;
      *(v8bf*)&Bs[nxt][dB + 8] = rb1;
    }
    __syncthreads();
  }

  // epilogue: +bias, exact GELU, store bf16
  const int colL   = lane & 15;
  const int rowOff = (lane < 16) ? 0 : 8;
  const float* b1e = b1 + expert * kF;
#pragma unroll
  for (int half = 0; half < 2; ++half) {
#pragma unroll
    for (int j = 0; j < 4; ++j) {
      const int n = n0 + wn * 64 + j * 16 + colL;
      const float bias = b1e[n];
#pragma unroll
      for (int r = 0; r < 8; ++r) {
        const int m = m0 + wm * 32 + half * 16 + rowOff + r;
        const float v = acc[half][j][r] + bias;
        const float g = 0.5f * v * (1.0f + erff(v * 0.70710678118654752440f));
        h[(size_t)m * kF + n] = f2bf(g);
      }
    }
  }
}

// ---------------------------------------------------------------------------
// FFN2 per expert: out[tok] += w * (h @ W2[e] + b2[e])
// ---------------------------------------------------------------------------
__global__ __launch_bounds__(256) void ffn2_kernel(const u16* __restrict__ hbuf,
                                                   const u16* __restrict__ w2s,
                                                   const float* __restrict__ b2,
                                                   const int* __restrict__ cnt,
                                                   const int* __restrict__ atok,
                                                   const float* __restrict__ awgt,
                                                   float* __restrict__ out, int expert) {
  const int count = cnt[expert];
  const int m0 = blockIdx.y * 128;
  if (m0 >= count) return;
  const int n0 = blockIdx.x * 128;

  __shared__ __align__(16) u16 As[2][128 * 32];
  __shared__ __align__(16) u16 Bs[2][4096];

  const int tid = threadIdx.x, lane = tid & 31, wave = tid >> 5;
  const int wm = wave >> 1, wn = wave & 1;

  const int ar = tid >> 2;
  const int ak = (tid & 3) * 8;
  const u16* pA0 = hbuf + (size_t)(m0 + ar) * kF + ak;
  const u16* pA1 = hbuf + (size_t)(m0 + ar + 64) * kF + ak;
  const int dA0 = ar * 32 + ak, dA1 = (ar + 64) * 32 + ak;

  const u16* pB = w2s + (((size_t)expert * (kF / 32)) * (kD / 16) + (n0 >> 4)) * 512 + tid * 16;
  const size_t bStep = (size_t)(kD / 16) * 512;
  const int dB = tid * 16;

  v8f acc[2][4] = {};

  v8bf ra0 = *(const v8bf*)pA0;
  v8bf ra1 = *(const v8bf*)pA1;
  v8bf rb0 = *(const v8bf*)pB;
  v8bf rb1 = *(const v8bf*)(pB + 8);
  *(v8bf*)&As[0][dA0] = ra0;
  *(v8bf*)&As[0][dA1] = ra1;
  *(v8bf*)&Bs[0][dB]     = rb0;
  *(v8bf*)&Bs[0][dB + 8] = rb1;
  __syncthreads();

  const int KSTEPS = kF / 32;
  for (int it = 0; it < KSTEPS; ++it) {
    const int cur = it & 1, nxt = cur ^ 1;
    if (it + 1 < KSTEPS) {
      pA0 += 32; pA1 += 32; pB += bStep;
      ra0 = *(const v8bf*)pA0;
      ra1 = *(const v8bf*)pA1;
      rb0 = *(const v8bf*)pB;
      rb1 = *(const v8bf*)(pB + 8);
    }
    const u16* Ac = As[cur];
    const u16* Bc = Bs[cur];
    const v16bf a0 = load_a_frag(Ac, wm * 32, lane);
    const v16bf a1 = load_a_frag(Ac, wm * 32 + 16, lane);
    v16bf bf[4];
#pragma unroll
    for (int j = 0; j < 4; ++j) bf[j] = load_b_frag(Bc, wn * 4 + j, lane);
#pragma unroll
    for (int j = 0; j < 4; ++j) {
      acc[0][j] = wmma_bf16(a0, bf[j], acc[0][j]);
      acc[1][j] = wmma_bf16(a1, bf[j], acc[1][j]);
    }
    if (it + 1 < KSTEPS) {
      *(v8bf*)&As[nxt][dA0] = ra0;
      *(v8bf*)&As[nxt][dA1] = ra1;
      *(v8bf*)&Bs[nxt][dB]     = rb0;
      *(v8bf*)&Bs[nxt][dB + 8] = rb1;
    }
    __syncthreads();
  }

  const int colL   = lane & 15;
  const int rowOff = (lane < 16) ? 0 : 8;
  const int* tok   = atok + expert * kN;
  const float* wgt = awgt + expert * kN;
  const float* b2e = b2 + expert * kD;
#pragma unroll
  for (int half = 0; half < 2; ++half) {
#pragma unroll
    for (int r = 0; r < 8; ++r) {
      const int m = m0 + wm * 32 + half * 16 + rowOff + r;
      if (m < count) {
        const int   t = tok[m];
        const float w = wgt[m];
        float* orow = out + (size_t)t * kD;
#pragma unroll
        for (int j = 0; j < 4; ++j) {
          const int n = n0 + wn * 64 + j * 16 + colL;
          orow[n] += w * (acc[half][j][r] + b2e[n]);
        }
      }
    }
  }
}

// ---------------------------------------------------------------------------
// aux_loss = sum_e( (cnt[e]/N) * (mean_n probs[n][e]) ) * E   (deterministic tree)
// ---------------------------------------------------------------------------
__global__ __launch_bounds__(256) void finalize_kernel(const float* __restrict__ probs_all,
                                                       const int* __restrict__ cnt,
                                                       float* __restrict__ aux_out) {
  __shared__ float red[256][kE];
  const int tid = threadIdx.x;
  float p[kE];
#pragma unroll
  for (int e = 0; e < kE; ++e) p[e] = 0.0f;
  for (int t = tid; t < kN; t += 256)
#pragma unroll
    for (int e = 0; e < kE; ++e) p[e] += probs_all[t * kE + e];
#pragma unroll
  for (int e = 0; e < kE; ++e) red[tid][e] = p[e];
  __syncthreads();
  for (int off = 128; off; off >>= 1) {
    if (tid < off)
#pragma unroll
      for (int e = 0; e < kE; ++e) red[tid][e] += red[tid + off][e];
    __syncthreads();
  }
  if (tid == 0) {
    float aux = 0.0f;
#pragma unroll
    for (int e = 0; e < kE; ++e)
      aux += ((float)cnt[e] * (1.0f / (float)kN)) * (red[0][e] * (1.0f / (float)kN));
    aux_out[0] = aux * (float)kE;
  }
}

// ---------------------------------------------------------------------------
// Workspace layout (bytes)
// ---------------------------------------------------------------------------
static constexpr size_t O_W1B  = 0;                       // 67,108,864  (swizzled bf16 W1)
static constexpr size_t O_W2B  = 67108864;                // 67,108,864  (swizzled bf16 W2)
static constexpr size_t O_XB   = 134217728;               // 16,777,216  (8192*1024 bf16)
static constexpr size_t O_H    = 150994944;               // 67,108,864  (8192*4096 bf16)
static constexpr size_t O_CNT  = 218103808;               // 256 (8 ints, padded)
static constexpr size_t O_TOK  = 218104064;               // 262,144 (8*8192 int)
static constexpr size_t O_WGT  = 218366208;               // 262,144 (8*8192 float)
static constexpr size_t O_PRB  = 218628352;               // 262,144 (8192*8 float)
static constexpr size_t WS_NEED = 218890496;

extern "C" void kernel_launch(void* const* d_in, const int* in_sizes, int n_in,
                              void* d_out, int out_size, void* d_ws, size_t ws_size,
                              hipStream_t stream) {
  (void)in_sizes; (void)n_in; (void)out_size;
  if (ws_size < WS_NEED) return;

  const float* x  = (const float*)d_in[0];
  const float* Wg = (const float*)d_in[1];
  const float* W1 = (const float*)d_in[2];
  const float* b1 = (const float*)d_in[3];
  const float* W2 = (const float*)d_in[4];
  const float* b2 = (const float*)d_in[5];
  float* out = (float*)d_out;

  char* ws = (char*)d_ws;
  u16*   w1s   = (u16*)(ws + O_W1B);
  u16*   w2s   = (u16*)(ws + O_W2B);
  u16*   xb    = (u16*)(ws + O_XB);
  u16*   hbuf  = (u16*)(ws + O_H);
  int*   cnt   = (int*)(ws + O_CNT);
  int*   atok  = (int*)(ws + O_TOK);
  float* awgt  = (float*)(ws + O_WGT);
  float* probs = (float*)(ws + O_PRB);

  hipMemsetAsync(cnt, 0, 256, stream);
  hipMemsetAsync(out, 0, (size_t)kN * kD * sizeof(float), stream);

  // weights -> bf16 fragment-swizzled (L2-resident at 134 MB total)
  swz_w_kernel<<<dim3(kF / 128, kD / 32, kE), 256, 0, stream>>>(W1, w1s, kF, kD / 32);
  swz_w_kernel<<<dim3(kD / 128, kF / 32, kE), 256, 0, stream>>>(W2, w2s, kD, kF / 32);
  cvt_bf16_kernel<<<(kN * kD) / 2048, 256, 0, stream>>>(x, xb, kN * kD);

  gate_kernel<<<kN / 8, 256, 0, stream>>>(x, Wg, cnt, atok, awgt, probs);

  const dim3 g1(kF / 128, kN / 128);   // 32 x 64
  const dim3 g2(kD / 128, kN / 128);   // 8  x 64
  for (int e = 0; e < kE; ++e) {
    ffn1_kernel<<<g1, 256, 0, stream>>>(xb, w1s, b1, cnt, atok, hbuf, e);
    ffn2_kernel<<<g2, 256, 0, stream>>>(hbuf, w2s, b2, cnt, atok, awgt, out, e);
  }

  finalize_kernel<<<1, 256, 0, stream>>>(probs, cnt, out + (size_t)kN * kD);
}
